// sim_body_90975997264410
// MI455X (gfx1250) — matrix operand order repaired and checked by reading, not compiled, not run on hardware
//
#include <hip/hip_runtime.h>
#include <math.h>

// ---------------------------------------------------------------------------
// Problem constants (B=4, S=2048, D=1024, K=4, A=2)
// ---------------------------------------------------------------------------
#define M_TOK 8192          // B*S tokens
#define DDIM  1024
#define FDIM  3072          // (A+1)*D
#define KEXP  4
#define XSLOT 2048          // column offset of x inside hcat = [link0|link1|x]
#define BSTRIDE 36          // LDS line stride (u16) for B slab: 64B data + 8B pad

typedef __bf16 v16bf __attribute__((ext_vector_type(16)));
typedef float  v8f   __attribute__((ext_vector_type(8)));
typedef unsigned int u32x4 __attribute__((ext_vector_type(4)));
typedef int          i32x8 __attribute__((ext_vector_type(8)));
typedef int          i32x4 __attribute__((ext_vector_type(4)));

union FragB16 { v16bf v; unsigned int u[8]; };

#if __has_builtin(__builtin_amdgcn_tensor_load_to_lds) && \
    __has_builtin(__builtin_amdgcn_s_wait_tensorcnt)
#define HAVE_TDM 1
#else
#define HAVE_TDM 0
#endif

__device__ __forceinline__ unsigned short f2bf(float f) {
  unsigned int u = __float_as_uint(f);
  u += 0x7FFFu + ((u >> 16) & 1u);       // round-to-nearest-even
  return (unsigned short)(u >> 16);
}

#if HAVE_TDM
// ---------------------------------------------------------------------------
// Tensor Data Mover: 2-D bf16 tile (xelems contiguous per line, ylines lines,
// line stride stride_elems) global -> LDS, optional LDS line padding.
// D# per CDNA5 ISA 8.3/8.4. pad_interval enc: 3 -> every 16 DWORDs (64B);
// pad_amount enc: 1 -> 2 DWORDs (8B).
// ---------------------------------------------------------------------------
__device__ __forceinline__ void tdm_load_2d(
    const unsigned short* gsrc, unsigned short* ldst,
    unsigned int xelems, unsigned int ylines, unsigned int stride_elems,
    unsigned int pen, unsigned int pint, unsigned int pamt) {
  unsigned long long ga = (unsigned long long)(uintptr_t)gsrc;
  unsigned int laddr = (unsigned int)(uintptr_t)ldst;   // low 32b = LDS offset
  u32x4 g0 = { 1u,                                      // count=1 (user D#)
               laddr,                                   // lds_addr [63:32]
               (unsigned int)ga,                        // global_addr [95:64]
               (unsigned int)((ga >> 32) & 0x01FFFFFFu) // global_addr [120:96]
                 | (2u << 30) };                        // type=2 ("image")
  i32x8 g1 = { (int)((1u << 16) | (pen << 20) |        // data_size=2B | pad
                     (pint << 22) | (pamt << 25)),
               (int)(xelems << 16),                     // tensor_dim0 [15:0]
               (int)((xelems >> 16) | (ylines << 16)),  // dim0[31:16] | dim1[15:0]
               (int)((ylines >> 16) | (xelems << 16)),  // dim1[31:16] | tile_dim0
               (int)ylines,                             // tile_dim1 (tile_dim2=0)
               (int)stride_elems,                       // tensor_dim0_stride lo32
               0, 0 };
  i32x4 z4 = { 0, 0, 0, 0 };
#if __clang_major__ >= 23
  i32x8 z8 = { 0, 0, 0, 0, 0, 0, 0, 0 };
  __builtin_amdgcn_tensor_load_to_lds(g0, g1, z4, z4, z8, 0);
#else
  __builtin_amdgcn_tensor_load_to_lds(g0, g1, z4, z4, 0);
#endif
}
#endif // HAVE_TDM

// ---------------------------------------------------------------------------
// Core: one block (8 waves / 256 thr) computes a 16-row x 1024-col f32 tile of
// A(bf16, row stride lda) x W (supplied TRANSPOSED: BT[n][Ktot], row stride
// ldb=Ktot). K-step = 32. TDM path: A tile (16x32) and the n-major B slab
// (1024 lines x 32, padded to 72B lines) are DMA'd into double-buffered LDS
// on TENSORcnt, wave 0 pipelining slab i+1 while all waves run WMMA on slab
// i. B fragments are pure ds_load_b64 (no VALU merge -> no WMMA hazard nops).
// acc[j] = wave's j-th 16x16 N-subtile, N base = waveId*128 + j*16.
// ---------------------------------------------------------------------------
__device__ __forceinline__ void gemm_rows16(
    const unsigned short* __restrict__ A, int lda, int Ktot,
    const unsigned short* __restrict__ BT,
    unsigned short* ldsA0, unsigned short* ldsA1,
    unsigned short* ldsB0, unsigned short* ldsB1,
    v8f* acc) {
  const int tid  = threadIdx.x;
  const int lane = tid & 31;
  const int wave = tid >> 5;

  #pragma unroll
  for (int j = 0; j < 8; ++j)
    #pragma unroll
    for (int v = 0; v < 8; ++v) acc[j][v] = 0.0f;

  const int rr  = lane & 15;
  const int kb  = (lane < 16) ? 0 : 8;   // unused: A and B fragments both use kb2
  const int cc  = lane & 15;
  const int kb2 = (lane < 16) ? 0 : 16;  // B K-base (16-bit B)

#if HAVE_TDM
  const int nsteps = Ktot / 32;
  if (wave == 0) {                       // prologue: slab 0 in flight
    tdm_load_2d(A, ldsA0, 32, 16, (unsigned)lda, 0, 0, 0);
    tdm_load_2d(BT, ldsB0, 32, 1024, (unsigned)Ktot, 1, 3, 1);
  }
  for (int i = 0; i < nsteps; ++i) {
    unsigned short* curA = (i & 1) ? ldsA1 : ldsA0;
    unsigned short* curB = (i & 1) ? ldsB1 : ldsB0;
    if (wave == 0) __builtin_amdgcn_s_wait_tensorcnt(0);  // slab i landed
    __syncthreads();   // slab i visible to all; slab i-1 readers all done
    if (wave == 0 && (i + 1) < nsteps) {                  // pipeline slab i+1
      int k0n = (i + 1) * 32;
      tdm_load_2d(A + k0n, (i & 1) ? ldsA0 : ldsA1, 32, 16, (unsigned)lda,
                  0, 0, 0);
      tdm_load_2d(BT + k0n, (i & 1) ? ldsB0 : ldsB1, 32, 1024,
                  (unsigned)Ktot, 1, 3, 1);
    }
    // A fragment: row-major [16][32] tile, dword d -> K pair
    FragB16 af;
    #pragma unroll
    for (int d = 0; d < 8; ++d) {
      int K0 = kb2 + (d << 1);
      af.u[d] = *(const unsigned int*)&curA[rr * 32 + K0];
    }
    // B fragments: n-major slab, 16 contiguous K per lane -> 4x ds_load_b64
    FragB16 bq[8];
    #pragma unroll
    for (int j = 0; j < 8; ++j) {
      const unsigned short* rowp =
          &curB[(wave * 128 + j * 16 + cc) * BSTRIDE + kb2];
      uint2 q0 = *(const uint2*)(rowp + 0);
      uint2 q1 = *(const uint2*)(rowp + 4);
      uint2 q2 = *(const uint2*)(rowp + 8);
      uint2 q3 = *(const uint2*)(rowp + 12);
      bq[j].u[0] = q0.x; bq[j].u[1] = q0.y;
      bq[j].u[2] = q1.x; bq[j].u[3] = q1.y;
      bq[j].u[4] = q2.x; bq[j].u[5] = q2.y;
      bq[j].u[6] = q3.x; bq[j].u[7] = q3.y;
    }
    #pragma unroll
    for (int j = 0; j < 8; ++j)
      acc[j] = __builtin_amdgcn_wmma_f32_16x16x32_bf16(
          false, af.v, false, bq[j].v, (short)0, acc[j], false, false);
  }
#else
  // ---------------- manual staging fallback (no TDM builtin) ---------------
  for (int k0 = 0; k0 < Ktot; k0 += 32) {
    __syncthreads();
    for (int e = tid; e < 16 * 32; e += 256) {
      int r = e >> 5, c = e & 31;
      ldsA0[r * 32 + c] = A[(size_t)r * lda + k0 + c];
    }
    for (int ln = tid; ln < 1024; ln += 256) {     // straight copy, Wt layout
      const uint2* s = (const uint2*)(BT + (size_t)ln * Ktot + k0);
      uint2* dp = (uint2*)&ldsB0[ln * BSTRIDE];
      dp[0] = s[0]; dp[1] = s[1]; dp[2] = s[2]; dp[3] = s[3];
    }
    if (k0 + 32 < Ktot)
      __builtin_prefetch(BT + (size_t)(tid * 4) * Ktot + k0 + 32, 0, 1);
    __syncthreads();
    FragB16 af;
    #pragma unroll
    for (int d = 0; d < 8; ++d) {
      int K0 = kb2 + (d << 1);
      af.u[d] = *(const unsigned int*)&ldsA0[rr * 32 + K0];
    }
    FragB16 bq[8];
    #pragma unroll
    for (int j = 0; j < 8; ++j) {
      const unsigned short* rowp =
          &ldsB0[(wave * 128 + j * 16 + cc) * BSTRIDE + kb2];
      uint2 q0 = *(const uint2*)(rowp + 0);
      uint2 q1 = *(const uint2*)(rowp + 4);
      uint2 q2 = *(const uint2*)(rowp + 8);
      uint2 q3 = *(const uint2*)(rowp + 12);
      bq[j].u[0] = q0.x; bq[j].u[1] = q0.y;
      bq[j].u[2] = q1.x; bq[j].u[3] = q1.y;
      bq[j].u[4] = q2.x; bq[j].u[5] = q2.y;
      bq[j].u[6] = q3.x; bq[j].u[7] = q3.y;
    }
    #pragma unroll
    for (int j = 0; j < 8; ++j)
      acc[j] = __builtin_amdgcn_wmma_f32_16x16x32_bf16(
          false, af.v, false, bq[j].v, (short)0, acc[j], false, false);
  }
#endif
}

#if HAVE_TDM
#define GEMM_LDS_DECL                                                  \
  __shared__ __align__(16) unsigned short ldsA0[16 * 32];              \
  __shared__ __align__(16) unsigned short ldsA1[16 * 32];              \
  __shared__ __align__(16) unsigned short ldsB0[1024 * BSTRIDE];       \
  __shared__ __align__(16) unsigned short ldsB1[1024 * BSTRIDE];
#define GEMM_LDS_ARGS ldsA0, ldsA1, ldsB0, ldsB1
#else
#define GEMM_LDS_DECL                                                  \
  __shared__ __align__(16) unsigned short ldsA0[16 * 32];              \
  __shared__ __align__(16) unsigned short ldsB0[1024 * BSTRIDE];
#define GEMM_LDS_ARGS ldsA0, ldsA0, ldsB0, ldsB0
#endif

// ---------------------------------------------------------------------------
// Stage 1 (x2): x <- x + LN(x @ Wm[c] + bm[c]); writes xf (f32) + hcat x-slot
// ---------------------------------------------------------------------------
__global__ void __launch_bounds__(256) mem_stage_kernel(
    float* __restrict__ xf, unsigned short* __restrict__ hcat,
    const unsigned short* __restrict__ WmT, const float* __restrict__ bm,
    const float* __restrict__ ln_g, const float* __restrict__ ln_b, int c) {
  GEMM_LDS_DECL
  __shared__ float sSum[16], sSq[16], sMean[16], sRstd[16];

  const int r0 = blockIdx.x * 16;
  if (threadIdx.x < 16) { sSum[threadIdx.x] = 0.0f; sSq[threadIdx.x] = 0.0f; }

  v8f acc[8];
  gemm_rows16(hcat + (size_t)r0 * FDIM + XSLOT, FDIM, DDIM,
              WmT + (size_t)c * DDIM * DDIM, GEMM_LDS_ARGS, acc);

  const int lane = threadIdx.x & 31;
  const int wave = threadIdx.x >> 5;
  const int half = lane >> 4;
  const int cbase = wave * 128 + (lane & 15);

  float psum[8], psq[8];
  #pragma unroll
  for (int v = 0; v < 8; ++v) { psum[v] = 0.0f; psq[v] = 0.0f; }
  #pragma unroll
  for (int j = 0; j < 8; ++j) {
    float bj = bm[c * DDIM + cbase + j * 16];
    #pragma unroll
    for (int v = 0; v < 8; ++v) {
      float t = acc[j][v] + bj;
      psum[v] += t; psq[v] += t * t;
    }
  }
  #pragma unroll
  for (int m = 8; m >= 1; m >>= 1)
    #pragma unroll
    for (int v = 0; v < 8; ++v) {
      psum[v] += __shfl_xor(psum[v], m, 32);
      psq[v]  += __shfl_xor(psq[v],  m, 32);
    }
  if ((lane & 15) == 0)
    #pragma unroll
    for (int v = 0; v < 8; ++v) {
      atomicAdd(&sSum[v + 8 * half], psum[v]);   // ds_add_f32
      atomicAdd(&sSq [v + 8 * half], psq[v]);
    }
  __syncthreads();
  if (threadIdx.x < 16) {
    float mean = sSum[threadIdx.x] * (1.0f / DDIM);
    float var  = sSq[threadIdx.x] * (1.0f / DDIM) - mean * mean;
    sMean[threadIdx.x] = mean;
    sRstd[threadIdx.x] = rsqrtf(var + 1e-5f);
  }
  __syncthreads();

  #pragma unroll
  for (int j = 0; j < 8; ++j) {
    int col = cbase + j * 16;
    float g = ln_g[col], b = ln_b[col], bj = bm[c * DDIM + col];
    #pragma unroll
    for (int v = 0; v < 8; ++v) {
      int rloc = v + 8 * half;
      size_t row = (size_t)(r0 + rloc);
      float mem = acc[j][v] + bj;
      float ln  = (mem - sMean[rloc]) * sRstd[rloc] * g + b;
      float xn  = xf[row * DDIM + col] + ln;
      xf[row * DDIM + col] = xn;
      hcat[row * FDIM + XSLOT + col] = f2bf(xn);
    }
  }
}

// ---------------------------------------------------------------------------
// Stage 2: gate logits + top-2 (first-occurrence max, matching lax.top_k)
// ---------------------------------------------------------------------------
__global__ void __launch_bounds__(256) gate_topk_kernel(
    const float* __restrict__ xf, const float* __restrict__ Wg,
    const float* __restrict__ bg, int* __restrict__ idx) {
  const int lane = threadIdx.x & 31;
  const int row  = blockIdx.x * 8 + (threadIdx.x >> 5);
  const float* xr = xf + (size_t)row * DDIM;
  float a0 = 0.f, a1 = 0.f, a2 = 0.f, a3 = 0.f;
  for (int i = lane; i < DDIM; i += 32) {
    float xv = xr[i];
    float4 w = *(const float4*)(Wg + (size_t)i * 4);
    a0 += xv * w.x; a1 += xv * w.y; a2 += xv * w.z; a3 += xv * w.w;
  }
  #pragma unroll
  for (int m = 16; m >= 1; m >>= 1) {
    a0 += __shfl_xor(a0, m, 32); a1 += __shfl_xor(a1, m, 32);
    a2 += __shfl_xor(a2, m, 32); a3 += __shfl_xor(a3, m, 32);
  }
  if (lane == 0) {
    float lg[4] = {a0 + bg[0], a1 + bg[1], a2 + bg[2], a3 + bg[3]};
    int i0 = 0;
    for (int k = 1; k < 4; ++k) if (lg[k] > lg[i0]) i0 = k;
    int i1 = -1;
    for (int k = 0; k < 4; ++k)
      if (k != i0 && (i1 < 0 || lg[k] > lg[i1])) i1 = k;
    idx[row * 2 + 0] = i0;
    idx[row * 2 + 1] = i1;
  }
}

// ---------------------------------------------------------------------------
// Stage 3: expert k GEMM, scatter selected rows (+be) into link slots of hcat
// ---------------------------------------------------------------------------
__global__ void __launch_bounds__(256) expert_stage_kernel(
    unsigned short* __restrict__ hcat, const unsigned short* __restrict__ WeT,
    const float* __restrict__ be, const int* __restrict__ idx, int kexp) {
  GEMM_LDS_DECL

  const int r0 = blockIdx.x * 16;
  v8f acc[8];
  gemm_rows16(hcat + (size_t)r0 * FDIM + XSLOT, FDIM, DDIM,
              WeT + (size_t)kexp * DDIM * DDIM, GEMM_LDS_ARGS, acc);

  const int lane = threadIdx.x & 31;
  const int wave = threadIdx.x >> 5;
  const int half = lane >> 4;
  const int cbase = wave * 128 + (lane & 15);

  #pragma unroll
  for (int v = 0; v < 8; ++v) {
    size_t row = (size_t)(r0 + v + 8 * half);
    int i0 = idx[row * 2 + 0], i1 = idx[row * 2 + 1];
    bool s0 = (i0 == kexp), s1 = (i1 == kexp);
    if (!s0 && !s1) continue;
    #pragma unroll
    for (int j = 0; j < 8; ++j) {
      int col = cbase + j * 16;
      unsigned short bv = f2bf(acc[j][v] + be[kexp * DDIM + col]);
      if (s0) hcat[row * FDIM + col] = bv;          // link slot 0
      if (s1) hcat[row * FDIM + DDIM + col] = bv;   // link slot 1
    }
  }
}

// ---------------------------------------------------------------------------
// Stage 4: h = [link|x] @ Wfit + bfit -> LN -> exact GELU -> out (f32)
// ---------------------------------------------------------------------------
__global__ void __launch_bounds__(256) fit_stage_kernel(
    const unsigned short* __restrict__ hcat, const unsigned short* __restrict__ WfT,
    const float* __restrict__ bfit, const float* __restrict__ ln_g,
    const float* __restrict__ ln_b, float* __restrict__ out) {
  GEMM_LDS_DECL
  __shared__ float sSum[16], sSq[16], sMean[16], sRstd[16];

  const int r0 = blockIdx.x * 16;
  if (threadIdx.x < 16) { sSum[threadIdx.x] = 0.0f; sSq[threadIdx.x] = 0.0f; }

  v8f acc[8];
  gemm_rows16(hcat + (size_t)r0 * FDIM, FDIM, FDIM, WfT, GEMM_LDS_ARGS, acc);

  const int lane = threadIdx.x & 31;
  const int wave = threadIdx.x >> 5;
  const int half = lane >> 4;
  const int cbase = wave * 128 + (lane & 15);

  float psum[8], psq[8];
  #pragma unroll
  for (int v = 0; v < 8; ++v) { psum[v] = 0.0f; psq[v] = 0.0f; }
  #pragma unroll
  for (int j = 0; j < 8; ++j) {
    float bj = bfit[cbase + j * 16];
    #pragma unroll
    for (int v = 0; v < 8; ++v) {
      float t = acc[j][v] + bj;
      psum[v] += t; psq[v] += t * t;
    }
  }
  #pragma unroll
  for (int m = 8; m >= 1; m >>= 1)
    #pragma unroll
    for (int v = 0; v < 8; ++v) {
      psum[v] += __shfl_xor(psum[v], m, 32);
      psq[v]  += __shfl_xor(psq[v],  m, 32);
    }
  if ((lane & 15) == 0)
    #pragma unroll
    for (int v = 0; v < 8; ++v) {
      atomicAdd(&sSum[v + 8 * half], psum[v]);
      atomicAdd(&sSq [v + 8 * half], psq[v]);
    }
  __syncthreads();
  if (threadIdx.x < 16) {
    float mean = sSum[threadIdx.x] * (1.0f / DDIM);
    float var  = sSq[threadIdx.x] * (1.0f / DDIM) - mean * mean;
    sMean[threadIdx.x] = mean;
    sRstd[threadIdx.x] = rsqrtf(var + 1e-5f);
  }
  __syncthreads();

  #pragma unroll
  for (int j = 0; j < 8; ++j) {
    int col = cbase + j * 16;
    float g = ln_g[col], b = ln_b[col], bj = bfit[col];
    #pragma unroll
    for (int v = 0; v < 8; ++v) {
      int rloc = v + 8 * half;
      size_t row = (size_t)(r0 + rloc);
      float h  = acc[j][v] + bj;
      float ln = (h - sMean[rloc]) * sRstd[rloc] * g + b;
      float ge = 0.5f * ln * (1.0f + erff(ln * 0.70710678118654752f));
      out[row * DDIM + col] = ge;
    }
  }
}

// ---------------------------------------------------------------------------
// Helpers: f32 -> bf16 TRANSPOSING weight conversion (dst[m][n][k] =
// bf16(src[m][k][n])), so GEMM B operands are n-major and TDM can deliver
// them directly in WMMA B register order; x init (f32 copy + bf16 into hcat)
// ---------------------------------------------------------------------------
__global__ void __launch_bounds__(256) cvt_bf16_t_kernel(
    const float* __restrict__ src, unsigned short* __restrict__ dst,
    int K, int N, int nmat) {
  int total = nmat * K * N;
  for (int i = blockIdx.x * 256 + threadIdx.x; i < total; i += gridDim.x * 256) {
    int m = i / (K * N);
    int r = i - m * (K * N);
    int k = r / N, n = r - k * N;
    dst[(size_t)m * K * N + (size_t)n * K + k] = f2bf(src[i]);
  }
}

__global__ void __launch_bounds__(256) init_x_kernel(
    const float* __restrict__ x, float* __restrict__ xf,
    unsigned short* __restrict__ hcat, int n) {
  for (int i = blockIdx.x * 256 + threadIdx.x; i < n; i += gridDim.x * 256) {
    float v = x[i];
    xf[i] = v;
    size_t row = (size_t)(i >> 10);
    int col = i & 1023;
    hcat[row * FDIM + XSLOT + col] = f2bf(v);
  }
}

// ---------------------------------------------------------------------------
// Launch
// ---------------------------------------------------------------------------
extern "C" void kernel_launch(void* const* d_in, const int* in_sizes, int n_in,
                              void* d_out, int out_size, void* d_ws, size_t ws_size,
                              hipStream_t stream) {
  const float* x    = (const float*)d_in[0];
  const float* Wm   = (const float*)d_in[1];
  const float* bm   = (const float*)d_in[2];
  const float* ln_g = (const float*)d_in[3];
  const float* ln_b = (const float*)d_in[4];
  const float* Wg   = (const float*)d_in[5];
  const float* bg   = (const float*)d_in[6];
  const float* We   = (const float*)d_in[7];
  const float* be   = (const float*)d_in[8];
  const float* Wfit = (const float*)d_in[9];
  const float* bfit = (const float*)d_in[10];
  float* out = (float*)d_out;

  // workspace carve-up (~99 MB total)
  char* ws = (char*)d_ws;
  float* xf = (float*)ws;                 ws += (size_t)M_TOK * DDIM * sizeof(float);
  unsigned short* hcat = (unsigned short*)ws; ws += (size_t)M_TOK * FDIM * 2;
  unsigned short* WmT  = (unsigned short*)ws; ws += (size_t)2 * DDIM * DDIM * 2;
  unsigned short* WeT  = (unsigned short*)ws; ws += (size_t)KEXP * DDIM * DDIM * 2;
  unsigned short* WfT  = (unsigned short*)ws; ws += (size_t)FDIM * DDIM * 2;
  int* idx = (int*)ws;                    ws += (size_t)M_TOK * 2 * sizeof(int);

  // one-time (per call) bf16 transposed weight conversion (lives in L2 after)
  cvt_bf16_t_kernel<<<2048, 256, 0, stream>>>(Wm,   WmT, DDIM, DDIM, 2);
  cvt_bf16_t_kernel<<<2048, 256, 0, stream>>>(We,   WeT, DDIM, DDIM, KEXP);
  cvt_bf16_t_kernel<<<2048, 256, 0, stream>>>(Wfit, WfT, FDIM, DDIM, 1);
  init_x_kernel    <<<4096, 256, 0, stream>>>(x, xf, hcat, M_TOK * DDIM);

  // sequential memory experts (stream order = dependency order)
  mem_stage_kernel<<<M_TOK / 16, 256, 0, stream>>>(xf, hcat, WmT, bm, ln_g, ln_b, 0);
  mem_stage_kernel<<<M_TOK / 16, 256, 0, stream>>>(xf, hcat, WmT, bm, ln_g, ln_b, 1);

  // gate + top-2
  gate_topk_kernel<<<M_TOK / 8, 256, 0, stream>>>(xf, Wg, bg, idx);

  // linker experts, scattered into hcat link slots
  for (int k = 0; k < KEXP; ++k)
    expert_stage_kernel<<<M_TOK / 16, 256, 0, stream>>>(hcat, WeT, be, idx, k);

  // fit + LN + GELU
  fit_stage_kernel<<<M_TOK / 16, 256, 0, stream>>>(hcat, WfT, bfit, ln_g, ln_b, out);
}
